// PointNet2seg_19000935317877
// MI455X (gfx1250) — matrix operand order repaired and checked by reading, not compiled, not run on hardware
//
#include <hip/hip_runtime.h>
#include <hip/hip_bf16.h>

typedef __attribute__((ext_vector_type(16))) _Float16 v16h;
typedef __attribute__((ext_vector_type(8)))  _Float16 h8;
typedef __attribute__((ext_vector_type(8)))  float    v8f;
typedef _Float16 half_t;

#define NSLAB 128
#define BN_EPS_C 1e-5f

static __host__ __device__ inline int padk(int ci) { return (ci + 31) & ~31; }

// ---------------------------------------------------------------------------
// input transpose: [B,6,N] -> xyz f32 [B,N,3], pts f16 [B,N,3], both f16 [B,N,6]
// ---------------------------------------------------------------------------
__global__ __launch_bounds__(256) void k_transpose_in(
    const float* __restrict__ in, float* __restrict__ xyz,
    half_t* __restrict__ pts, half_t* __restrict__ both, int B, int N) {
  int i = blockIdx.x * 256 + threadIdx.x;
  int total = B * N * 6;
  if (i >= total) return;
  int n = i % N;
  int c = (i / N) % 6;
  int b = i / (N * 6);
  float v = in[i];
  both[((size_t)b * N + n) * 6 + c] = (half_t)v;
  if (c < 3) xyz[((size_t)b * N + n) * 3 + c] = v;
  else       pts[((size_t)b * N + n) * 3 + (c - 3)] = (half_t)v;
}

// ---------------------------------------------------------------------------
// farthest point sampling: one block (1024 thr) per batch, xyz+dist in LDS
// ---------------------------------------------------------------------------
__global__ __launch_bounds__(1024) void k_fps(
    const float* __restrict__ xyz, int* __restrict__ fi, int N, int npoint) {
  __shared__ float sx[2048 * 3];
  __shared__ float dist[2048];
  __shared__ float rv[1024];
  __shared__ int   ri[1024];
  int b = blockIdx.x, t = threadIdx.x;
  for (int n = t; n < N; n += 1024) {
    sx[n * 3 + 0] = xyz[((size_t)b * N + n) * 3 + 0];
    sx[n * 3 + 1] = xyz[((size_t)b * N + n) * 3 + 1];
    sx[n * 3 + 2] = xyz[((size_t)b * N + n) * 3 + 2];
    dist[n] = 1e10f;
  }
  __syncthreads();
  int far = 0;
  for (int it = 0; it < npoint; ++it) {
    if (t == 0) fi[b * npoint + it] = far;
    float cx = sx[far * 3], cy = sx[far * 3 + 1], cz = sx[far * 3 + 2];
    float bv = -1e30f; int bi = 0x7fffffff;
    for (int n = t; n < N; n += 1024) {
      float dx = sx[n * 3] - cx, dy = sx[n * 3 + 1] - cy, dz = sx[n * 3 + 2] - cz;
      float d = dx * dx + dy * dy + dz * dz;
      float dn = dist[n];
      if (d < dn) { dn = d; dist[n] = d; }
      if (dn > bv || (dn == bv && n < bi)) { bv = dn; bi = n; }
    }
    rv[t] = bv; ri[t] = bi;
    __syncthreads();
    for (int s = 512; s > 0; s >>= 1) {
      if (t < s) {
        float v2 = rv[t + s]; int i2 = ri[t + s];
        if (v2 > rv[t] || (v2 == rv[t] && i2 < ri[t])) { rv[t] = v2; ri[t] = i2; }
      }
      __syncthreads();
    }
    far = ri[0];
    __syncthreads();   // everyone read ri[0] before it is overwritten
  }
}

__global__ __launch_bounds__(256) void k_gather_xyz(
    const float* __restrict__ xyz, const int* __restrict__ fi,
    float* __restrict__ out, int B, int N, int S) {
  int i = blockIdx.x * 256 + threadIdx.x;
  if (i >= B * S * 3) return;
  int c = i % 3, s = (i / 3) % S, b = i / (3 * S);
  out[i] = xyz[((size_t)b * N + fi[b * S + s]) * 3 + c];
}

// ---------------------------------------------------------------------------
// ball query: first K in-radius indices in ascending order, padded with first
// ---------------------------------------------------------------------------
__global__ __launch_bounds__(256) void k_ballquery(
    const float* __restrict__ xyz, const float* __restrict__ cent,
    int* __restrict__ gi, int B, int N, int S, int K, float r2) {
  int i = blockIdx.x * 256 + threadIdx.x;
  if (i >= B * S) return;
  int b = i / S;
  float cx = cent[i * 3], cy = cent[i * 3 + 1], cz = cent[i * 3 + 2];
  long base = (long)i * K;
  int cnt = 0;
  const float* xp = xyz + (size_t)b * N * 3;
  for (int n = 0; n < N && cnt < K; ++n) {
    float dx = xp[n * 3] - cx, dy = xp[n * 3 + 1] - cy, dz = xp[n * 3 + 2] - cz;
    if (dx * dx + dy * dy + dz * dz <= r2) gi[base + cnt++] = n;
  }
  int f = (cnt > 0) ? gi[base] : 0;
  for (int j = cnt; j < K; ++j) gi[base + j] = f;
}

// ---------------------------------------------------------------------------
// grouping + concat -> padded f16 X:  [xyz[gi]-center (3) | pts[gi] (D) | 0-pad]
// ---------------------------------------------------------------------------
__global__ __launch_bounds__(256) void k_group(
    const float* __restrict__ xyz, const half_t* __restrict__ pts,
    const float* __restrict__ cent, const int* __restrict__ gi,
    half_t* __restrict__ X, int B, int N, int S, int K, int D, int Cp) {
  int C = 3 + D;
  long total = (long)B * S * K * Cp;
  long i = (long)blockIdx.x * 256 + threadIdx.x;
  if (i >= total) return;
  int c = (int)(i % Cp);
  long r = i / Cp;
  if (c >= C) { X[i] = (half_t)0.f; return; }
  int k = (int)(r % K);
  long bs = r / K;
  int b = (int)(bs / S);
  int p = gi[bs * K + k];
  if (c < 3) X[i] = (half_t)(xyz[((size_t)b * N + p) * 3 + c] - cent[bs * 3 + c]);
  else       X[i] = pts[((size_t)b * N + p) * D + (c - 3)];
}

// weight convert+pad: W f32 [Co,Ci] -> W16 f16 [Co,CiP], pad zeros
__global__ __launch_bounds__(256) void k_wcvt(
    const float* __restrict__ W, half_t* __restrict__ W16,
    int Co, int Ci, int CiP) {
  long i = (long)blockIdx.x * 256 + threadIdx.x;
  long total = (long)Co * CiP;
  if (i >= total) return;
  int c = (int)(i % CiP);
  long o = i / CiP;
  W16[i] = (c < Ci) ? (half_t)W[o * Ci + c] : (half_t)0.f;
}

// ---------------------------------------------------------------------------
// WMMA GEMM:  Y[M,Co] = X16[M,CiP] * W16[Co,CiP]^T + bias   (CiP % 32 == 0)
// one wave per 16x16 tile; aligned vector loads, no tail, no in-loop cvt
// A lane(m=lane%16,h=lane/16): k = 16h+e
// B lane(n=lane%16,h=lane/16): k = 16h+e
// D lane elem r -> (m=r+8h, n=lane%16)
// ---------------------------------------------------------------------------
__global__ __launch_bounds__(256) void k_mm_wmma(
    const half_t* __restrict__ X, const half_t* __restrict__ W,
    const float* __restrict__ bias, float* __restrict__ Y,
    int M, int Ci, int Co) {
  int wave = threadIdx.x >> 5, lane = threadIdx.x & 31;
  int half = lane >> 4, l16 = lane & 15;
  int n0 = blockIdx.x << 4;
  int m0 = (blockIdx.y * 8 + wave) << 4;
  const half_t* xrow = X + (size_t)(m0 + l16) * Ci;
  const half_t* wrow = W + (size_t)(n0 + l16) * Ci;
  v8f acc = {0.f, 0.f, 0.f, 0.f, 0.f, 0.f, 0.f, 0.f};
  for (int k0 = 0; k0 < Ci; k0 += 32) {
    h8 alo = *(const h8*)(xrow + k0 + half * 16);       // 16B aligned
    h8 ahi = *(const h8*)(xrow + k0 + half * 16 + 8);   // 16B aligned
    v16h a = __builtin_shufflevector(alo, ahi, 0, 1, 2, 3, 4, 5, 6, 7,
                                     8, 9, 10, 11, 12, 13, 14, 15);
    v16h bm = *(const v16h*)(wrow + k0 + half * 16);    // 32B aligned
    acc = __builtin_amdgcn_wmma_f32_16x16x32_f16(false, a, false, bm,
                                                 (short)0, acc, false, false);
  }
  float bb = bias[n0 + l16];
  size_t col = (size_t)n0 + l16;
#pragma unroll
  for (int r = 0; r < 8; ++r) {
    int m = m0 + r + half * 8;
    Y[(size_t)m * Co + col] = acc[r] + bb;
  }
}

// ---------------------------------------------------------------------------
// deterministic batchnorm stats: slab partials -> finalize -> apply(+relu->f16)
// ---------------------------------------------------------------------------
__global__ __launch_bounds__(256) void k_bn_partial(
    const float* __restrict__ Y, float* __restrict__ part, int M, int Co) {
  int slab = blockIdx.x, t = threadIdx.x;
  int mb = (int)((long)M * slab / NSLAB);
  int me = (int)((long)M * (slab + 1) / NSLAB);
  float s[4] = {0.f, 0.f, 0.f, 0.f}, q[4] = {0.f, 0.f, 0.f, 0.f};
  for (int m = mb; m < me; ++m) {
    const float* row = Y + (size_t)m * Co;
#pragma unroll
    for (int j = 0; j < 4; ++j) {
      int c = t + j * 256;
      if (c < Co) { float v = row[c]; s[j] += v; q[j] += v * v; }
    }
  }
#pragma unroll
  for (int j = 0; j < 4; ++j) {
    int c = t + j * 256;
    if (c < Co) {
      part[((size_t)slab * Co + c) * 2 + 0] = s[j];
      part[((size_t)slab * Co + c) * 2 + 1] = q[j];
    }
  }
}

__global__ __launch_bounds__(256) void k_bn_finalize(
    const float* __restrict__ part, const float* __restrict__ g,
    const float* __restrict__ be, float* __restrict__ scale,
    float* __restrict__ shift, int M, int Co) {
  int c = blockIdx.x * 256 + threadIdx.x;
  if (c >= Co) return;
  float s = 0.f, q = 0.f;
  for (int sl = 0; sl < NSLAB; ++sl) {
    s += part[((size_t)sl * Co + c) * 2 + 0];
    q += part[((size_t)sl * Co + c) * 2 + 1];
  }
  float mean = s / (float)M;
  float var  = q / (float)M - mean * mean;
  float sc = g[c] * rsqrtf(var + BN_EPS_C);
  scale[c] = sc;
  shift[c] = be[c] - mean * sc;
}

__global__ __launch_bounds__(256) void k_bn_apply(
    const float* __restrict__ Y, const float* __restrict__ scale,
    const float* __restrict__ shift, half_t* __restrict__ out,
    long total, int Co) {
  long i = (long)blockIdx.x * 256 + threadIdx.x;
  if (i >= total) return;
  int c = (int)(i % Co);
  float v = Y[i] * scale[c] + shift[c];
  out[i] = (half_t)(v > 0.f ? v : 0.f);
}

// ---------------------------------------------------------------------------
// max pool over K samples: A f16 [outer,K,C] -> out f16 [outer, Cout] col coff
// ---------------------------------------------------------------------------
__global__ __launch_bounds__(256) void k_maxpool(
    const half_t* __restrict__ A, half_t* __restrict__ out,
    int outer, int K, int C, int Cout, int coff) {
  long i = (long)blockIdx.x * 256 + threadIdx.x;
  long total = (long)outer * C;
  if (i >= total) return;
  int c = (int)(i % C);
  long o = i / C;
  const half_t* p = A + (size_t)o * K * C + c;
  float m = (float)p[0];
  for (int k = 1; k < K; ++k) { float v = (float)p[(size_t)k * C]; m = v > m ? v : m; }
  out[o * Cout + coff + c] = (half_t)m;
}

// concat [xyz(3) | pts(D) | pad]  for group-all stage
__global__ __launch_bounds__(256) void k_concat_sa3(
    const float* __restrict__ xyz, const half_t* __restrict__ pts,
    half_t* __restrict__ X, int B, int S, int D, int Cp) {
  int C = 3 + D;
  long total = (long)B * S * Cp;
  long i = (long)blockIdx.x * 256 + threadIdx.x;
  if (i >= total) return;
  int c = (int)(i % Cp);
  long r = i / Cp;
  if (c >= C) { X[i] = (half_t)0.f; return; }
  X[i] = (c < 3) ? (half_t)xyz[r * 3 + c] : pts[(size_t)r * D + (c - 3)];
}

// concat [p1(C1) | broadcast p3(C2) | pad]  for fp3 (S2==1)
__global__ __launch_bounds__(256) void k_concat_fp3(
    const half_t* __restrict__ p1, const half_t* __restrict__ p3,
    half_t* __restrict__ X, int B, int S, int C1, int C2, int Cp) {
  int C = C1 + C2;
  long total = (long)B * S * Cp;
  long i = (long)blockIdx.x * 256 + threadIdx.x;
  if (i >= total) return;
  int c = (int)(i % Cp);
  long r = i / Cp;
  if (c >= C) { X[i] = (half_t)0.f; return; }
  int b = (int)(r / S);
  X[i] = (c < C1) ? p1[(size_t)r * C1 + c] : p3[(size_t)b * C2 + (c - C1)];
}

// 3-NN (squared distance, stable ties), inverse-distance weights
__global__ __launch_bounds__(256) void k_knn3(
    const float* __restrict__ q, const float* __restrict__ ref,
    int* __restrict__ idx, float* __restrict__ w, int B, int N, int S) {
  int i = blockIdx.x * 256 + threadIdx.x;
  if (i >= B * N) return;
  int b = i / N;
  float qx = q[i * 3], qy = q[i * 3 + 1], qz = q[i * 3 + 2];
  const float* rp = ref + (size_t)b * S * 3;
  float d0 = 1e30f, d1 = 1e30f, d2 = 1e30f;
  int i0 = 0, i1 = 0, i2 = 0;
  for (int s = 0; s < S; ++s) {
    float dx = rp[s * 3] - qx, dy = rp[s * 3 + 1] - qy, dz = rp[s * 3 + 2] - qz;
    float d = dx * dx + dy * dy + dz * dz;
    if (d < d0)      { d2 = d1; i2 = i1; d1 = d0; i1 = i0; d0 = d; i0 = s; }
    else if (d < d1) { d2 = d1; i2 = i1; d1 = d;  i1 = s; }
    else if (d < d2) { d2 = d;  i2 = s; }
  }
  float r0 = 1.f / (d0 + 1e-8f), r1 = 1.f / (d1 + 1e-8f), r2 = 1.f / (d2 + 1e-8f);
  float rs = r0 + r1 + r2;
  w[i * 3] = r0 / rs; w[i * 3 + 1] = r1 / rs; w[i * 3 + 2] = r2 / rs;
  idx[i * 3] = i0; idx[i * 3 + 1] = i1; idx[i * 3 + 2] = i2;
}

// concat [p1(C1) | interp(p2)(C2) | pad] using precomputed 3-NN idx/weights
__global__ __launch_bounds__(256) void k_concat_fp(
    const half_t* __restrict__ p1, const half_t* __restrict__ p2,
    const int* __restrict__ idx, const float* __restrict__ w,
    half_t* __restrict__ X, int B, int N, int S, int C1, int C2, int Cp) {
  int C = C1 + C2;
  long total = (long)B * N * Cp;
  long i = (long)blockIdx.x * 256 + threadIdx.x;
  if (i >= total) return;
  int c = (int)(i % Cp);
  long r = i / Cp;
  if (c >= C) { X[i] = (half_t)0.f; return; }
  int b = (int)(r / N);
  if (c < C1) { X[i] = p1[(size_t)r * C1 + c]; return; }
  int o = c - C1;
  long k3 = r * 3;
  const half_t* pb = p2 + (size_t)b * S * C2;
  float v = w[k3 + 0] * (float)pb[(size_t)idx[k3 + 0] * C2 + o]
          + w[k3 + 1] * (float)pb[(size_t)idx[k3 + 1] * C2 + o]
          + w[k3 + 2] * (float)pb[(size_t)idx[k3 + 2] * C2 + o];
  X[i] = (half_t)v;
}

// head: 128 -> 3 (too narrow for WMMA tiles); plain dot products
__global__ __launch_bounds__(256) void k_head_mm(
    const half_t* __restrict__ X, const float* __restrict__ W,
    const float* __restrict__ b, float* __restrict__ Y, int M, int Ci) {
  int i = blockIdx.x * 256 + threadIdx.x;
  if (i >= M) return;
  const half_t* x = X + (size_t)i * Ci;
#pragma unroll
  for (int o = 0; o < 3; ++o) {
    const float* wr = W + o * Ci;
    float s = b[o];
    for (int c = 0; c < Ci; ++c) s += (float)x[c] * wr[c];
    Y[(size_t)i * 3 + o] = s;
  }
}

__global__ __launch_bounds__(256) void k_head_out(
    const float* __restrict__ Y, const float* __restrict__ scale,
    const float* __restrict__ shift, float* __restrict__ out, int B, int N) {
  int i = blockIdx.x * 256 + threadIdx.x;
  if (i >= B * N) return;
  int b = i / N, n = i % N;
#pragma unroll
  for (int o = 0; o < 3; ++o) {
    float v = Y[(size_t)i * 3 + o] * scale[o] + shift[o];
    v = v > 0.f ? v : 0.f;
    out[((size_t)b * 3 + o) * N + n] = v;
  }
}

// ---------------------------------------------------------------------------
// host orchestration
// ---------------------------------------------------------------------------
struct LayerP { const float *W, *b, *g, *be; };

extern "C" void kernel_launch(void* const* d_in, const int* in_sizes, int n_in,
                              void* d_out, int out_size, void* d_ws, size_t ws_size,
                              hipStream_t stream) {
  (void)out_size; (void)ws_size;
  // locate inputs by unique element counts (robust to flatten order)
  int xyz_i = -1, sa1 = -1, sa2 = -1, sa3 = -1, fp3 = -1, fp2 = -1, fp1 = -1, hd = -1;
  for (int i = 0; i < n_in; ++i) {
    switch (in_sizes[i]) {
      case 98304:   if (xyz_i < 0) xyz_i = i; break;  // xyz [8,6,2048]
      case 768:     sa1 = i; break;                   // W[128,6]
      case 66304:   sa2 = i; break;                   // W[256,259]
      case 263680:  sa3 = i; break;                   // W[512,515]
      case 1572864: fp3 = i; break;                   // W[1024,1536]
      case 393216:  fp2 = i; break;                   // W[512,768]
      case 67072:   fp1 = i; break;                   // W[256,262]
      case 384:     hd  = i; break;                   // W[3,128]
      default: break;
    }
  }
  if (xyz_i < 0 || sa1 < 0 || sa2 < 0 || sa3 < 0 || fp3 < 0 || fp2 < 0 || fp1 < 0 || hd < 0)
    return;
  auto L = [&](int base) -> LayerP {
    return { (const float*)d_in[base], (const float*)d_in[base + 1],
             (const float*)d_in[base + 2], (const float*)d_in[base + 3] };
  };
  const float* in_xyz = (const float*)d_in[xyz_i];
  float* out = (float*)d_out;

  // workspace arena
  char* p = (char*)d_ws;
  auto alloc = [&](size_t bytes) -> void* {
    char* r = p; p += (bytes + 255) & ~(size_t)255; return (void*)r;
  };
  const int B = 8, N0 = 2048, S1 = 1024, K1 = 32, S2 = 512, K2 = 64;
  const long MG = 262144;  // B*S1*K1 == B*S2*K2
  half_t* bufA   = (half_t*)alloc((size_t)MG * 288 * 2);   // padded f16 activations
  half_t* bufB   = (half_t*)alloc((size_t)MG * 256 * 2);
  float*  bufY   = (float*)alloc((size_t)MG * 256 * 4);    // pre-BN f32
  half_t* w16    = (half_t*)alloc((size_t)1024 * 1536 * 2);
  float*  l0_xyz = (float*)alloc((size_t)B * N0 * 3 * 4);
  half_t* l0_pts = (half_t*)alloc((size_t)B * N0 * 3 * 2);
  half_t* l0_bth = (half_t*)alloc((size_t)B * N0 * 6 * 2);
  float*  l1_xyz = (float*)alloc((size_t)B * S1 * 3 * 4);
  float*  l2_xyz = (float*)alloc((size_t)B * S2 * 3 * 4);
  half_t* l1_pts = (half_t*)alloc((size_t)B * S1 * 256 * 2);
  half_t* l2_pts = (half_t*)alloc((size_t)B * S2 * 512 * 2);
  half_t* l2f    = (half_t*)alloc((size_t)B * S2 * 512 * 2);
  half_t* l1f    = (half_t*)alloc((size_t)B * S1 * 256 * 2);
  half_t* l3_pts = (half_t*)alloc((size_t)B * 1024 * 2);
  float*  heady  = (float*)alloc((size_t)B * N0 * 3 * 4);
  float*  part   = (float*)alloc((size_t)NSLAB * 1024 * 2 * 4);
  float*  scale  = (float*)alloc(1024 * 4);
  float*  shift  = (float*)alloc(1024 * 4);
  int*    fi1    = (int*)alloc((size_t)B * S1 * 4);
  int*    fi2    = (int*)alloc((size_t)B * S2 * 4);
  int*    gi1    = (int*)alloc((size_t)B * S1 * K1 * 4);
  int*    gi2    = (int*)alloc((size_t)B * S2 * K2 * 4);
  int*    ki2    = (int*)alloc((size_t)B * S1 * 3 * 4);
  float*  kw2    = (float*)alloc((size_t)B * S1 * 3 * 4);
  int*    ki1    = (int*)alloc((size_t)B * N0 * 3 * 4);
  float*  kw1    = (float*)alloc((size_t)B * N0 * 3 * 4);

  auto gblk = [](long n) { return (unsigned)((n + 255) / 256); };
  // X is padded-f16 [M, padk(Ci)]; writes actOut f16 [M, Co]
  auto mlp = [&](const half_t* X, int M, int Ci, int Co, LayerP ly, half_t* actOut) {
    int CiP = padk(Ci);
    k_wcvt<<<gblk((long)Co * CiP), 256, 0, stream>>>(ly.W, w16, Co, Ci, CiP);
    dim3 g(Co / 16, M / 128);
    k_mm_wmma<<<g, 256, 0, stream>>>(X, w16, ly.b, bufY, M, CiP, Co);
    k_bn_partial<<<NSLAB, 256, 0, stream>>>(bufY, part, M, Co);
    k_bn_finalize<<<(Co + 255) / 256, 256, 0, stream>>>(part, ly.g, ly.be,
                                                        scale, shift, M, Co);
    long total = (long)M * Co;
    k_bn_apply<<<gblk(total), 256, 0, stream>>>(bufY, scale, shift, actOut, total, Co);
  };

  // ---- input transpose ----
  k_transpose_in<<<gblk((long)B * N0 * 6), 256, 0, stream>>>(in_xyz, l0_xyz, l0_pts,
                                                             l0_bth, B, N0);
  // ---- SA1 ----
  k_fps<<<B, 1024, 0, stream>>>(l0_xyz, fi1, N0, S1);
  k_gather_xyz<<<gblk((long)B * S1 * 3), 256, 0, stream>>>(l0_xyz, fi1, l1_xyz, B, N0, S1);
  k_ballquery<<<gblk((long)B * S1), 256, 0, stream>>>(l0_xyz, l1_xyz, gi1, B, N0, S1,
                                                      K1, 0.04f);
  k_group<<<gblk(MG * 32), 256, 0, stream>>>(l0_xyz, l0_pts, l1_xyz, gi1, bufA,
                                             B, N0, S1, K1, 3, 32);
  mlp(bufA, (int)MG, 6, 128, L(sa1), bufB);
  mlp(bufB, (int)MG, 128, 256, L(sa1 + 4), bufA);
  k_maxpool<<<gblk((long)B * S1 * 256), 256, 0, stream>>>(bufA, l1_pts, B * S1, K1,
                                                          256, 256, 0);
  // ---- SA2 ----
  k_fps<<<B, 1024, 0, stream>>>(l1_xyz, fi2, S1, S2);
  k_gather_xyz<<<gblk((long)B * S2 * 3), 256, 0, stream>>>(l1_xyz, fi2, l2_xyz, B, S1, S2);
  k_ballquery<<<gblk((long)B * S2), 256, 0, stream>>>(l1_xyz, l2_xyz, gi2, B, S1, S2,
                                                      K2, 0.16f);
  k_group<<<gblk(MG * 288), 256, 0, stream>>>(l1_xyz, l1_pts, l2_xyz, gi2, bufA,
                                              B, S1, S2, K2, 256, 288);
  mlp(bufA, (int)MG, 259, 256, L(sa2), bufB);
  {
    LayerP l2l = L(sa2 + 4);  // W [512,256]; split into 2 x 256-channel chunks
    for (int ch = 0; ch < 2; ++ch) {
      LayerP lc = { l2l.W + (size_t)ch * 256 * 256, l2l.b + ch * 256,
                    l2l.g + ch * 256, l2l.be + ch * 256 };
      mlp(bufB, (int)MG, 256, 256, lc, bufA);
      k_maxpool<<<gblk((long)B * S2 * 256), 256, 0, stream>>>(bufA, l2_pts, B * S2,
                                                              K2, 256, 512, ch * 256);
    }
  }
  // ---- SA3 (group all) ----
  const int M3 = B * S2;  // 4096
  k_concat_sa3<<<gblk((long)M3 * 544), 256, 0, stream>>>(l2_xyz, l2_pts, bufA,
                                                         B, S2, 512, 544);
  mlp(bufA, M3, 515, 512, L(sa3), bufB);
  mlp(bufB, M3, 512, 1024, L(sa3 + 4), bufA);
  k_maxpool<<<gblk((long)B * 1024), 256, 0, stream>>>(bufA, l3_pts, B, S2, 1024, 1024, 0);
  // ---- FP3 (broadcast interp) ----
  k_concat_fp3<<<gblk((long)M3 * 1536), 256, 0, stream>>>(l2_pts, l3_pts, bufB, B, S2,
                                                          512, 1024, 1536);
  mlp(bufB, M3, 1536, 1024, L(fp3), bufA);
  mlp(bufA, M3, 1024, 512, L(fp3 + 4), l2f);
  // ---- FP2 ----
  k_knn3<<<gblk((long)B * S1), 256, 0, stream>>>(l1_xyz, l2_xyz, ki2, kw2, B, S1, S2);
  k_concat_fp<<<gblk((long)B * S1 * 768), 256, 0, stream>>>(l1_pts, l2f, ki2, kw2, bufA,
                                                            B, S1, S2, 256, 512, 768);
  mlp(bufA, B * S1, 768, 512, L(fp2), bufB);
  mlp(bufB, B * S1, 512, 256, L(fp2 + 4), l1f);
  // ---- FP1 ----
  k_knn3<<<gblk((long)B * N0), 256, 0, stream>>>(l0_xyz, l1_xyz, ki1, kw1, B, N0, S1);
  k_concat_fp<<<gblk((long)B * N0 * 288), 256, 0, stream>>>(l0_bth, l1f, ki1, kw1, bufA,
                                                            B, N0, S1, 6, 256, 288);
  mlp(bufA, B * N0, 262, 256, L(fp1), bufB);
  mlp(bufB, B * N0, 256, 128, L(fp1 + 4), bufA);
  // ---- head ----
  LayerP lh = L(hd);
  k_head_mm<<<gblk((long)B * N0), 256, 0, stream>>>(bufA, lh.W, lh.b, heady, B * N0, 128);
  k_bn_partial<<<NSLAB, 256, 0, stream>>>(heady, part, B * N0, 3);
  k_bn_finalize<<<1, 256, 0, stream>>>(part, lh.g, lh.be, scale, shift, B * N0, 3);
  k_head_out<<<gblk((long)B * N0), 256, 0, stream>>>(heady, scale, shift, out, B, N0);
}
